// SpectralDynamicsEmbedding_38749194944604
// MI455X (gfx1250) — compile-verified, matrix operand order checked by reading, not run on hardware
//
#include <hip/hip_runtime.h>
#include <hip/hip_bf16.h>

// ---------------------------------------------------------------------------
// SpectralDynamicsEmbedding for MI455X (gfx1250, wave32, WMMA 16x16x32 f16)
// All DFTs / mixing / channel GEMMs as v_wmma_f32_16x16x32_f16 tiles.
// All B operands stored transposed (NxK) so A- and B-fragment loads are the
// same two contiguous 16B vector loads per lane (ds_load_b128 / b128 global).
// ---------------------------------------------------------------------------

typedef _Float16 h16;
typedef __attribute__((ext_vector_type(16))) _Float16 v16h;
typedef __attribute__((ext_vector_type(8)))  _Float16 v8h;
typedef __attribute__((ext_vector_type(8)))  float    v8f;

#define B_  8
#define H_  256
#define W_  256
#define C_  64

__device__ __forceinline__ int lane_id() { return threadIdx.x & 31; }

// Fragment load (16x32 f16 MxK for A, or NxK for transposed B — identical
// addressing): lane l -> row (l&15), K-groups {kb..kb+7, kb+16..kb+23},
// kb = (l>>4)*8.  Two aligned 16B vector loads + shuffle, no scalar packing.
__device__ __forceinline__ v16h frag16(const h16* q) {
  v8h lo = *(const v8h*)(q);
  v8h hi = *(const v8h*)(q + 16);
  return __builtin_shufflevector(lo, hi, 0, 1, 2, 3, 4, 5, 6, 7,
                                 8, 9, 10, 11, 12, 13, 14, 15);
}
__device__ __forceinline__ v16h ldA(const h16* p, int ld) {
  int l = lane_id();
  return frag16(p + (size_t)(l & 15) * ld + ((l >> 4) << 3));
}
__device__ __forceinline__ v16h ldA_neg(const h16* p, int ld) {
  return -ldA(p, ld);  // for the -Ai*Bi term (f16 WMMA has no A-neg modifier)
}

__device__ __forceinline__ v8f wmma_step(v16h a, v16h b, v8f c) {
  return __builtin_amdgcn_wmma_f32_16x16x32_f16(false, a, false, b, (short)0, c, false, false);
}

// D-tile stores. Layout: VGPR r -> M=r (lanes 0-15) / M=r+8 (lanes 16-31), N=lane&15.
__device__ __forceinline__ void stD_f32(float* p, int ld, v8f c) {
  int l = lane_id();
  int n = l & 15, mo = (l >> 4) << 3;
#pragma unroll
  for (int r = 0; r < 8; ++r) p[(size_t)(mo + r) * ld + n] = c[r];
}
__device__ __forceinline__ void stD_h16(h16* p, int ld, v8f c) {
  int l = lane_id();
  int n = l & 15, mo = (l >> 4) << 3;
#pragma unroll
  for (int r = 0; r < 8; ++r) p[(size_t)(mo + r) * ld + n] = (h16)c[r];
}
// Generalized strided store (element (m,n) -> p[m*ms + n*ns])
__device__ __forceinline__ void stD_h16s(h16* p, size_t ms, size_t ns, v8f c) {
  int l = lane_id();
  int n = l & 15, mo = (l >> 4) << 3;
#pragma unroll
  for (int r = 0; r < 8; ++r) p[(size_t)(mo + r) * ms + (size_t)n * ns] = (h16)c[r];
}
// Same, only rows m<8 valid (padded-batch tiles)
__device__ __forceinline__ void stD_h16s_guard8(h16* p, size_t ms, size_t ns, v8f c) {
  int l = lane_id();
  int n = l & 15, mo = (l >> 4) << 3;
  if (mo == 0) {
#pragma unroll
    for (int r = 0; r < 8; ++r) p[(size_t)r * ms + (size_t)n * ns] = (h16)c[r];
  }
}

// ---------------------------------------------------------------------------
__global__ void k_zero(float* p, int n) {
  int i = blockIdx.x * blockDim.x + threadIdx.x;
  if (i < n) p[i] = 0.f;
}

// ---------------------------------------------------------------------------
// K1: DFT twiddle tables (f16)
// ---------------------------------------------------------------------------
__global__ void k_tables(h16* twf_c, h16* twf_s, h16* thf_r, h16* thf_i,
                         h16* thi_r, h16* thi_i, h16* twi_a, h16* twi_b) {
  int tid = blockIdx.x * blockDim.x + threadIdx.x;  // 32768 threads
  const float PI2 = 6.28318530717958647692f;
  if (tid < 64 * 256) {  // forward W (truncated to 64 modes)
    int y = tid >> 8, w = tid & 255;
    float ang = PI2 * (float)((w * y) & 255) * (1.f / 256.f);
    twf_c[tid] = (h16)__cosf(ang);
    twf_s[tid] = (h16)(-__sinf(ang));
  }
  if (tid < 128 * 256) {  // forward H rows 0..63 and 192..255
    int xi = tid >> 8, hh = tid & 255;
    int xa = (xi < 64) ? xi : xi + 128;
    float ang = PI2 * (float)((hh * xa) & 255) * (1.f / 256.f);
    thf_r[tid] = (h16)__cosf(ang);
    thf_i[tid] = (h16)(-__sinf(ang));
  }
  if (tid < 256 * 128) {  // inverse H: (1/256) e^{+i...}
    int hh = tid >> 7, xi = tid & 127;
    int xa = (xi < 64) ? xi : xi + 128;
    float ang = PI2 * (float)((hh * xa) & 255) * (1.f / 256.f);
    thi_r[tid] = (h16)(__cosf(ang) * (1.f / 256.f));
    thi_i[tid] = (h16)(__sinf(ang) * (1.f / 256.f));
  }
  if (tid < 256 * 64) {  // inverse real W: s_y/256 * {cos, -sin}
    int w = tid >> 6, y = tid & 63;
    float s = ((y == 0) ? 1.f : 2.f) * (1.f / 256.f);
    float ang = PI2 * (float)((w * y) & 255) * (1.f / 256.f);
    twi_a[tid] = (h16)(__cosf(ang) * s);
    twi_b[tid] = (h16)(-__sinf(ang) * s);
  }
}

// ---------------------------------------------------------------------------
// K2: forward DFT along W.  out Y layout: [b][y][c][h]  (B-transposed for K3)
// ---------------------------------------------------------------------------
__global__ void k_fft_w(const float* __restrict__ u, const h16* __restrict__ twf_c,
                        const h16* __restrict__ twf_s, h16* __restrict__ Yr,
                        h16* __restrict__ Yi) {
  __shared__ __attribute__((aligned(16))) h16 uST[64][72];  // c x w-chunk (transposed B)
  int h = blockIdx.x, b = blockIdx.y;
  int t = threadIdx.x, wave = t >> 5;
  const float* ub = u + (size_t)(b * H_ + h) * W_ * C_;
  v8f acc[4];
#pragma unroll
  for (int q = 0; q < 4; ++q) acc[q] = {};
  for (int k0 = 0; k0 < 256; k0 += 64) {
    __syncthreads();
    for (int i = t; i < 64 * 64; i += 256) {
      int w = i >> 6, c = i & 63;  // consecutive t -> consecutive c: coalesced
      uST[c][w] = (h16)ub[(size_t)(k0 + w) * 64 + c];
    }
    if (k0 + 64 < 256) __builtin_prefetch(ub + (size_t)(k0 + 64) * 64, 0, 0);
    __syncthreads();
#pragma unroll
    for (int q = 0; q < 4; ++q) {
      int j = wave * 4 + q;  // 32 tile-jobs: 16 Yr(cos) + 16 Yi(sin)
      int part = j >> 4, pos = j & 15;
      int m0 = (pos >> 2) * 16, n0 = (pos & 3) * 16;
      const h16* A = (part ? twf_s : twf_c) + m0 * 256 + k0;
      for (int kk = 0; kk < 64; kk += 32)
        acc[q] = wmma_step(ldA(A + kk, 256), ldA(&uST[n0][kk], 72), acc[q]);
    }
  }
#pragma unroll
  for (int q = 0; q < 4; ++q) {
    int j = wave * 4 + q;
    int part = j >> 4, pos = j & 15;
    int m0 = (pos >> 2) * 16, n0 = (pos & 3) * 16;
    // element (m=y, n=c) -> Y[(((b*64+y)*64+c)*256 + h]
    h16* out = (part ? Yi : Yr) + (((size_t)b * 64 + m0) * 64 + n0) * 256 + h;
    stD_h16s(out, (size_t)64 * 256, 256, acc[q]);
  }
}

// ---------------------------------------------------------------------------
// K3: forward DFT along H (complex, 128 rows).  out Z layout: [x][y][b][c]
// (row-major A operand for K4's batch-mixing)
// ---------------------------------------------------------------------------
__global__ void k_fft_h(const h16* __restrict__ Yr, const h16* __restrict__ Yi,
                        const h16* __restrict__ thf_r, const h16* __restrict__ thf_i,
                        h16* __restrict__ Zr, h16* __restrict__ Zi) {
  __shared__ __attribute__((aligned(16))) h16 BrT[64][72], BiT[64][72];  // c x h-chunk
  int y = blockIdx.x, b = blockIdx.y;
  int t = threadIdx.x, wave = t >> 5;
  const h16* yrT = Yr + ((size_t)(b * 64 + y) * 64) * 256;  // rows c, cols h
  const h16* yiT = Yi + ((size_t)(b * 64 + y) * 64) * 256;
  v8f acc[8];
#pragma unroll
  for (int q = 0; q < 8; ++q) acc[q] = {};
  for (int k0 = 0; k0 < 256; k0 += 64) {
    __syncthreads();
    for (int i = t; i < 64 * 8; i += 256) {  // v8h-granular staging
      int r = i >> 3, s = (i & 7) * 8;
      *(v8h*)&BrT[r][s] = *(const v8h*)(yrT + (size_t)r * 256 + k0 + s);
      *(v8h*)&BiT[r][s] = *(const v8h*)(yiT + (size_t)r * 256 + k0 + s);
    }
    if (k0 + 64 < 256) {
      __builtin_prefetch(yrT + (size_t)(t & 63) * 256 + k0 + 64, 0, 0);
      __builtin_prefetch(yiT + (size_t)(t & 63) * 256 + k0 + 64, 0, 0);
    }
    __syncthreads();
#pragma unroll
    for (int q = 0; q < 8; ++q) {
      int j = wave * 8 + q;  // 64 jobs: 32 tiles x {r,i}
      int part = j >> 5, pos = j & 31;
      int m0 = (pos >> 2) * 16, n0 = (pos & 3) * 16;
      const h16* Ar = thf_r + m0 * 256 + k0;
      const h16* Ai = thf_i + m0 * 256 + k0;
      for (int kk = 0; kk < 64; kk += 32) {
        if (part == 0) {  // Zr = Ar*Br - Ai*Bi
          acc[q] = wmma_step(ldA(Ar + kk, 256), ldA(&BrT[n0][kk], 72), acc[q]);
          acc[q] = wmma_step(ldA_neg(Ai + kk, 256), ldA(&BiT[n0][kk], 72), acc[q]);
        } else {          // Zi = Ar*Bi + Ai*Br
          acc[q] = wmma_step(ldA(Ar + kk, 256), ldA(&BiT[n0][kk], 72), acc[q]);
          acc[q] = wmma_step(ldA(Ai + kk, 256), ldA(&BrT[n0][kk], 72), acc[q]);
        }
      }
    }
  }
#pragma unroll
  for (int q = 0; q < 8; ++q) {
    int j = wave * 8 + q;
    int part = j >> 5, pos = j & 31;
    int m0 = (pos >> 2) * 16, n0 = (pos & 3) * 16;
    // element (m=x, n=c) -> Z[(((x*64+y)*8+b)*64 + c]
    h16* out = (part ? Zi : Zr) + (((size_t)m0 * 64 + y) * 8 + b) * 64 + n0;
    stD_h16s(out, (size_t)64 * 8 * 64, 1, acc[q]);
  }
}

// ---------------------------------------------------------------------------
// K4: per-mode complex channel mixing (batch rows padded 8->16)
// out Zm layout: [b][y][c][x]  (B-transposed for K5)
// ---------------------------------------------------------------------------
__global__ void k_mix(const h16* __restrict__ Zr, const h16* __restrict__ Zi,
                      const float* __restrict__ wpr, const float* __restrict__ wpi,
                      const float* __restrict__ wnr, const float* __restrict__ wni,
                      h16* __restrict__ Zmr, h16* __restrict__ Zmi) {
  __shared__ __attribute__((aligned(16))) h16 ArS[16][72], AiS[16][72];  // b x i
  __shared__ __attribute__((aligned(16))) h16 BrT[64][72], BiT[64][72];  // o x i
  int xw = blockIdx.x, y = blockIdx.y, half = blockIdx.z;
  int xi = xw + half * 64;
  int t = threadIdx.x, wave = t >> 5;
  const float* wr = half ? wnr : wpr;
  const float* wi = half ? wni : wpi;
  // A: rows m=b (contiguous c in Z layout [x][y][b][c]); rows 8..15 zero
  {
    size_t base = ((size_t)xi * 64 + y) * 8 * 64;
    int i = t;  // 128 v8h units total (16 rows x 8)
    if (i < 16 * 8) {
      int m = i >> 3, s = (i & 7) * 8;
      v8h vr = {}, vi2 = {};
      if (m < 8) {
        vr = *(const v8h*)(Zr + base + (size_t)m * 64 + s);
        vi2 = *(const v8h*)(Zi + base + (size_t)m * 64 + s);
      }
      *(v8h*)&ArS[m][s] = vr;
      *(v8h*)&AiS[m][s] = vi2;
    }
  }
  // B^T[o][i] from scattered weights w[i][o][x][y]
  for (int i2 = t; i2 < 64 * 64; i2 += 256) {
    int ii = i2 >> 6, o = i2 & 63;
    size_t idx = (((size_t)ii * 64 + o) * 64 + xw) * 64 + y;
    BrT[o][ii] = (h16)wr[idx];
    BiT[o][ii] = (h16)wi[idx];
  }
  __syncthreads();
  int part = wave >> 2;  // 0: real out, 1: imag out
  int n0 = (wave & 3) * 16;
  v8f acc = {};
  for (int kk = 0; kk < 64; kk += 32) {
    if (part == 0) {
      acc = wmma_step(ldA(&ArS[0][kk], 72), ldA(&BrT[n0][kk], 72), acc);
      acc = wmma_step(-ldA(&AiS[0][kk], 72), ldA(&BiT[n0][kk], 72), acc);
    } else {
      acc = wmma_step(ldA(&ArS[0][kk], 72), ldA(&BiT[n0][kk], 72), acc);
      acc = wmma_step(ldA(&AiS[0][kk], 72), ldA(&BrT[n0][kk], 72), acc);
    }
  }
  // element (m=b, n=o) -> Zm[(((b*64+y)*64+o)*128 + xi]
  h16* out = (part ? Zmi : Zmr) + ((size_t)y * 64 + n0) * 128 + xi;
  stD_h16s_guard8(out, (size_t)64 * 64 * 128, 128, acc);
}

// ---------------------------------------------------------------------------
// K5: inverse DFT along H.  out XH layout: [b][h][c][y]  (B-transposed for K6)
// ---------------------------------------------------------------------------
__global__ void k_ifft_h(const h16* __restrict__ Zmr, const h16* __restrict__ Zmi,
                         const h16* __restrict__ thi_r, const h16* __restrict__ thi_i,
                         h16* __restrict__ XHr, h16* __restrict__ XHi) {
  __shared__ __attribute__((aligned(16))) h16 BrT[64][72], BiT[64][72];  // c x x-chunk
  int y = blockIdx.x, b = blockIdx.y, part = blockIdx.z;
  int t = threadIdx.x, wave = t >> 5;
  size_t base = ((size_t)(b * 64 + y) * 64) * 128;  // rows c, cols x (128)
  v8f acc[8];
#pragma unroll
  for (int q = 0; q < 8; ++q) acc[q] = {};
  for (int k0 = 0; k0 < 128; k0 += 64) {
    __syncthreads();
    for (int i = t; i < 64 * 8; i += 256) {
      int r = i >> 3, s = (i & 7) * 8;
      *(v8h*)&BrT[r][s] = *(const v8h*)(Zmr + base + (size_t)r * 128 + k0 + s);
      *(v8h*)&BiT[r][s] = *(const v8h*)(Zmi + base + (size_t)r * 128 + k0 + s);
    }
    __syncthreads();
#pragma unroll
    for (int q = 0; q < 8; ++q) {
      int j = wave * 8 + q;  // 64 tiles: 16(h) x 4(c)
      int m0 = (j >> 2) * 16, n0 = (j & 3) * 16;
      const h16* Ar = thi_r + m0 * 128 + k0;
      const h16* Ai = thi_i + m0 * 128 + k0;
      for (int kk = 0; kk < 64; kk += 32) {
        if (part == 0) {
          acc[q] = wmma_step(ldA(Ar + kk, 128), ldA(&BrT[n0][kk], 72), acc[q]);
          acc[q] = wmma_step(ldA_neg(Ai + kk, 128), ldA(&BiT[n0][kk], 72), acc[q]);
        } else {
          acc[q] = wmma_step(ldA(Ar + kk, 128), ldA(&BiT[n0][kk], 72), acc[q]);
          acc[q] = wmma_step(ldA(Ai + kk, 128), ldA(&BrT[n0][kk], 72), acc[q]);
        }
      }
    }
  }
  h16* Xout = part ? XHi : XHr;
#pragma unroll
  for (int q = 0; q < 8; ++q) {
    int j = wave * 8 + q;
    int m0 = (j >> 2) * 16, n0 = (j & 3) * 16;
    // element (m=h, n=c) -> XH[(((b*256+h)*64+c)*64 + y]
    h16* out = Xout + (((size_t)b * 256 + m0) * 64 + n0) * 64 + y;
    stD_h16s(out, (size_t)64 * 64, 64, acc[q]);
  }
}

// ---------------------------------------------------------------------------
// K6: fused inverse-DFT-W + LayerNorm + w_feat GEMM + w_v GEMM + sums
// One block per (b,h); W in two 128-row halves; LDS exactly 64 KB.
// ---------------------------------------------------------------------------
__global__ void k_spatial(const h16* __restrict__ XHr, const h16* __restrict__ XHi,
                          const h16* __restrict__ twi_a, const h16* __restrict__ twi_b,
                          const float* __restrict__ ln_g, const float* __restrict__ ln_b,
                          const float* __restrict__ w_feat, const float* __restrict__ w_v,
                          float* __restrict__ SumX, float* __restrict__ SumY,
                          float* __restrict__ vout) {
  __shared__ __attribute__((aligned(16))) h16 XrT[64][64], XiT[64][64];  // c x y
  __shared__ __attribute__((aligned(16))) h16 wfS[64][64], wvS[64][64];  // o x c (=B^T)
  __shared__ __attribute__((aligned(16))) h16 xA[128][64];               // w x c
  __shared__ __attribute__((aligned(16))) h16 xB[128][64];               // w x o
  int h = blockIdx.x, b = blockIdx.y;
  int t = threadIdx.x, wave = t >> 5;
  {
    const h16* xrT = XHr + ((size_t)(b * 256 + h) * 64) * 64;  // rows c, cols y
    const h16* xiT = XHi + ((size_t)(b * 256 + h) * 64) * 64;
    for (int i = t; i < 64 * 8; i += 256) {
      int r = i >> 3, s = (i & 7) * 8;
      *(v8h*)&XrT[r][s] = *(const v8h*)(xrT + (size_t)r * 64 + s);
      *(v8h*)&XiT[r][s] = *(const v8h*)(xiT + (size_t)r * 64 + s);
      // weights [o][c] f32 -> f16, same (row-major) layout == transposed-B
      const float* pf = w_feat + (size_t)r * 64 + s;
      const float* pv = w_v + (size_t)r * 64 + s;
      v8h hf, hv;
#pragma unroll
      for (int e = 0; e < 8; ++e) { hf[e] = (h16)pf[e]; hv[e] = (h16)pv[e]; }
      *(v8h*)&wfS[r][s] = hf;
      *(v8h*)&wvS[r][s] = hv;
    }
  }
  __syncthreads();
  for (int half = 0; half < 2; ++half) {
    int w0 = half * 128;
    // ---- x[w][c] = twi_a*XHr + twi_b*XHi ----
#pragma unroll
    for (int q = 0; q < 4; ++q) {
      int j = wave * 4 + q;  // 32 tiles: 8(w) x 4(c)
      int m0 = (j >> 2) * 16, n0 = (j & 3) * 16;
      const h16* Aa = twi_a + (size_t)(w0 + m0) * 64;
      const h16* Ab = twi_b + (size_t)(w0 + m0) * 64;
      v8f acc = {};
      for (int kk = 0; kk < 64; kk += 32) {
        acc = wmma_step(ldA(Aa + kk, 64), ldA(&XrT[n0][kk], 64), acc);
        acc = wmma_step(ldA(Ab + kk, 64), ldA(&XiT[n0][kk], 64), acc);
      }
      stD_h16(&xA[m0][n0], 64, acc);
    }
    __syncthreads();
    // ---- LayerNorm over c (vectorized row reads) ----
    if (t < 128) {
      const v8h* row = (const v8h*)&xA[t][0];
      float m = 0.f;
#pragma unroll
      for (int jv = 0; jv < 8; ++jv) {
        v8h v = row[jv];
#pragma unroll
        for (int e = 0; e < 8; ++e) m += (float)v[e];
      }
      m *= (1.f / 64.f);
      float var = 0.f;
#pragma unroll
      for (int jv = 0; jv < 8; ++jv) {
        v8h v = row[jv];
#pragma unroll
        for (int e = 0; e < 8; ++e) { float d = (float)v[e] - m; var += d * d; }
      }
      var *= (1.f / 64.f);
      float rs = rsqrtf(var + 1e-5f);
#pragma unroll
      for (int jv = 0; jv < 8; ++jv) {
        v8h v = row[jv], o;
#pragma unroll
        for (int e = 0; e < 8; ++e)
          o[e] = (h16)(((float)v[e] - m) * rs * ln_g[jv * 8 + e] + ln_b[jv * 8 + e]);
        ((v8h*)&xA[t][0])[jv] = o;
      }
    }
    __syncthreads();
    // ---- feat GEMM: xB = LN(x) @ w_feat^T ----
#pragma unroll
    for (int q = 0; q < 4; ++q) {
      int j = wave * 4 + q;
      int m0 = (j >> 2) * 16, n0 = (j & 3) * 16;
      v8f acc = {};
      for (int kk = 0; kk < 64; kk += 32)
        acc = wmma_step(ldA(&xA[m0][kk], 64), ldA(&wfS[n0][kk], 64), acc);
      stD_h16(&xB[m0][n0], 64, acc);
    }
    __syncthreads();
    // ---- v GEMM: v = xB @ w_v^T -> global ----
    float* vb = vout + ((size_t)(b * 256 + h) * 256 + w0) * 64;
#pragma unroll
    for (int q = 0; q < 4; ++q) {
      int j = wave * 4 + q;
      int m0 = (j >> 2) * 16, n0 = (j & 3) * 16;
      v8f acc = {};
      for (int kk = 0; kk < 64; kk += 32)
        acc = wmma_step(ldA(&xB[m0][kk], 64), ldA(&wvS[n0][kk], 64), acc);
      stD_f32(vb + (size_t)m0 * 64 + n0, 64, acc);
    }
    // ---- row/col sums of xB for the heads ----
    {
      int o = t & 63, p = t >> 6;  // 4 groups of 32 w-rows
      float s = 0.f;
      for (int w2 = p * 32; w2 < p * 32 + 32; ++w2) {
        float val = (float)xB[w2][o];
        s += val;
        atomicAdd(&SumY[((size_t)b * 256 + (w0 + w2)) * 64 + o], val);
      }
      atomicAdd(&SumX[((size_t)b * 256 + h) * 64 + o], s);
    }
    __syncthreads();
  }
}

// ---------------------------------------------------------------------------
// K7: head MLPs: out = LN(mean) @ w1^T @ w2^T   (ux: grid.y=0, uy: grid.y=1)
// ---------------------------------------------------------------------------
__global__ void k_head(const float* __restrict__ SumX, const float* __restrict__ SumY,
                       const float* __restrict__ gx, const float* __restrict__ bx,
                       const float* __restrict__ w1x, const float* __restrict__ w2x,
                       const float* __restrict__ gy, const float* __restrict__ by,
                       const float* __restrict__ w1y, const float* __restrict__ w2y,
                       float* __restrict__ outx, float* __restrict__ outy) {
  __shared__ __attribute__((aligned(16))) h16 aS[128][72], tS[128][72];
  __shared__ __attribute__((aligned(16))) h16 w1S[64][72], w2S[64][72];  // o x c (=B^T)
  int blk = blockIdx.x, which = blockIdx.y;
  const float* Sum = which ? SumY : SumX;
  const float* g = which ? gy : gx;
  const float* bb = which ? by : bx;
  const float* w1 = which ? w1y : w1x;
  const float* w2 = which ? w2y : w2x;
  float* out = which ? outy : outx;
  int t = threadIdx.x, wave = t >> 5;
  int R0 = blk * 128;
  for (int i = t; i < 64 * 8; i += 256) {
    int r = i >> 3, s = (i & 7) * 8;
    const float* p1 = w1 + (size_t)r * 64 + s;
    const float* p2 = w2 + (size_t)r * 64 + s;
    v8h h1, h2;
#pragma unroll
    for (int e = 0; e < 8; ++e) { h1[e] = (h16)p1[e]; h2[e] = (h16)p2[e]; }
    *(v8h*)&w1S[r][s] = h1;
    *(v8h*)&w2S[r][s] = h2;
  }
  if (t < 128) {
    const float* row = Sum + (size_t)(R0 + t) * 64;
    const float inv = 1.f / 256.f;  // mean over the reduced spatial axis
    float m = 0.f;
    for (int c = 0; c < 64; ++c) m += row[c];
    m *= inv * (1.f / 64.f);
    float var = 0.f;
    for (int c = 0; c < 64; ++c) { float d = row[c] * inv - m; var += d * d; }
    var *= (1.f / 64.f);
    float rs = rsqrtf(var + 1e-5f);
    for (int c = 0; c < 64; ++c)
      aS[t][c] = (h16)((row[c] * inv - m) * rs * g[c] + bb[c]);
  }
  __syncthreads();
#pragma unroll
  for (int q = 0; q < 4; ++q) {
    int j = wave * 4 + q;
    int m0 = (j >> 2) * 16, n0 = (j & 3) * 16;
    v8f acc = {};
    for (int kk = 0; kk < 64; kk += 32)
      acc = wmma_step(ldA(&aS[m0][kk], 72), ldA(&w1S[n0][kk], 72), acc);
    stD_h16(&tS[m0][n0], 72, acc);
  }
  __syncthreads();
#pragma unroll
  for (int q = 0; q < 4; ++q) {
    int j = wave * 4 + q;
    int m0 = (j >> 2) * 16, n0 = (j & 3) * 16;
    v8f acc = {};
    for (int kk = 0; kk < 64; kk += 32)
      acc = wmma_step(ldA(&tS[m0][kk], 72), ldA(&w2S[n0][kk], 72), acc);
    stD_f32(out + (size_t)(R0 + m0) * 64 + n0, 64, acc);
  }
}

// ---------------------------------------------------------------------------
// Launch
// ---------------------------------------------------------------------------
extern "C" void kernel_launch(void* const* d_in, const int* in_sizes, int n_in,
                              void* d_out, int out_size, void* d_ws, size_t ws_size,
                              hipStream_t stream) {
  (void)in_sizes; (void)n_in; (void)out_size; (void)ws_size;
  const float* u    = (const float*)d_in[0];
  const float* wpr  = (const float*)d_in[1];
  const float* wpi  = (const float*)d_in[2];
  const float* wnr  = (const float*)d_in[3];
  const float* wni  = (const float*)d_in[4];
  const float* ln_g = (const float*)d_in[5];
  const float* ln_b = (const float*)d_in[6];
  const float* w_feat = (const float*)d_in[7];
  const float* lnx_g = (const float*)d_in[8];
  const float* lnx_b = (const float*)d_in[9];
  const float* wx1 = (const float*)d_in[10];
  const float* wx2 = (const float*)d_in[11];
  const float* lny_g = (const float*)d_in[12];
  const float* lny_b = (const float*)d_in[13];
  const float* wy1 = (const float*)d_in[14];
  const float* wy2 = (const float*)d_in[15];
  const float* w_v = (const float*)d_in[16];

  float* out_ux = (float*)d_out;                  // [8,256,64]
  float* out_uy = out_ux + (size_t)B_ * H_ * C_;  // [8,256,64]
  float* out_v  = out_uy + (size_t)B_ * W_ * C_;  // [8,256,256,64]

  char* ws = (char*)d_ws;
  size_t off = 0;
  auto carve_h = [&](size_t nelem) { h16* p = (h16*)(ws + off); off += nelem * sizeof(h16); return p; };
  h16* twf_c = carve_h(64 * 256);
  h16* twf_s = carve_h(64 * 256);
  h16* thf_r = carve_h(128 * 256);
  h16* thf_i = carve_h(128 * 256);
  h16* thi_r = carve_h(256 * 128);
  h16* thi_i = carve_h(256 * 128);
  h16* twi_a = carve_h(256 * 64);
  h16* twi_b = carve_h(256 * 64);
  h16* Yr  = carve_h((size_t)B_ * 64 * 64 * 256);   // [b][y][c][h]
  h16* Yi  = carve_h((size_t)B_ * 64 * 64 * 256);
  h16* Zr  = carve_h((size_t)128 * 64 * B_ * 64);   // [x][y][b][c]
  h16* Zi  = carve_h((size_t)128 * 64 * B_ * 64);
  h16* Zmr = carve_h((size_t)B_ * 64 * 64 * 128);   // [b][y][c][x]
  h16* Zmi = carve_h((size_t)B_ * 64 * 64 * 128);
  h16* XHr = carve_h((size_t)B_ * 256 * 64 * 64);   // [b][h][c][y]
  h16* XHi = carve_h((size_t)B_ * 256 * 64 * 64);
  float* SumX = (float*)(ws + off); off += (size_t)B_ * H_ * C_ * sizeof(float);
  float* SumY = (float*)(ws + off); off += (size_t)B_ * W_ * C_ * sizeof(float);

  k_zero<<<dim3((2 * B_ * H_ * C_ + 255) / 256), dim3(256), 0, stream>>>(SumX, 2 * B_ * H_ * C_);
  k_tables<<<dim3(128), dim3(256), 0, stream>>>(twf_c, twf_s, thf_r, thf_i, thi_r, thi_i, twi_a, twi_b);
  k_fft_w<<<dim3(H_, B_), dim3(256), 0, stream>>>(u, twf_c, twf_s, Yr, Yi);
  k_fft_h<<<dim3(64, B_), dim3(256), 0, stream>>>(Yr, Yi, thf_r, thf_i, Zr, Zi);
  k_mix<<<dim3(64, 64, 2), dim3(256), 0, stream>>>(Zr, Zi, wpr, wpi, wnr, wni, Zmr, Zmi);
  k_ifft_h<<<dim3(64, B_, 2), dim3(256), 0, stream>>>(Zmr, Zmi, thi_r, thi_i, XHr, XHi);
  k_spatial<<<dim3(H_, B_), dim3(256), 0, stream>>>(XHr, XHi, twi_a, twi_b, ln_g, ln_b,
                                                    w_feat, w_v, SumX, SumY, out_v);
  k_head<<<dim3(16, 2), dim3(256), 0, stream>>>(SumX, SumY, lnx_g, lnx_b, wx1, wx2,
                                                lny_g, lny_b, wy1, wy2, out_ux, out_uy);
}
